// Diffusion_22144851378254
// MI455X (gfx1250) — matrix operand order repaired and checked by reading, not compiled, not run on hardware
//
#include <hip/hip_runtime.h>
#include <hip/hip_bf16.h>
#include <math.h>

typedef __bf16 bf16;
typedef __attribute__((ext_vector_type(16))) __bf16 v16bf;
typedef __attribute__((ext_vector_type(8)))  __bf16 v8bf;
typedef __attribute__((ext_vector_type(8)))  float  v8f;

typedef int i32x4 __attribute__((vector_size(16)));
typedef __attribute__((address_space(1))) void  as1_void;
typedef __attribute__((address_space(3))) void  as3_void;
typedef __attribute__((address_space(1))) i32x4 as1_i32x4;
typedef __attribute__((address_space(3))) i32x4 as3_i32x4;

#define BATCH 8192
#define DIM   2048
#define MROWS (BATCH*4)

#if __has_builtin(__builtin_amdgcn_global_load_async_to_lds_b128)
#define USE_ASYNC_LDS 1
#else
#define USE_ASYNC_LDS 0
#endif

__device__ __forceinline__ void wait_async0() {
#if __has_builtin(__builtin_amdgcn_s_wait_asynccnt)
  __builtin_amdgcn_s_wait_asynccnt(0);
#else
  asm volatile("s_wait_asynccnt 0" ::: "memory");
#endif
}

// ---------------------------------------------------------------------------
// K1: q_sample + timestep embedding + conds, LayerNorm over D, emit bf16 f
//     one block per (b, s) token row; 256 threads x 8 elements
// ---------------------------------------------------------------------------
__global__ __launch_bounds__(256) void k_build_tokens(
    const float* __restrict__ x_start, const int* __restrict__ t,
    const float* __restrict__ noise, const float* __restrict__ cond_src,
    const float* __restrict__ cond_tgt, const float* __restrict__ sqrt_ac,
    const float* __restrict__ sqrt_omac,
    bf16* __restrict__ f, float* __restrict__ maskf)
{
  const int bs  = blockIdx.x;   // b*4 + s
  const int s   = bs & 3;
  const int b   = bs >> 2;
  const int tid = threadIdx.x;

  float vals[8];
  float sum = 0.f, sumsq = 0.f, sumabs = 0.f;

  float sa = 0.f, so = 0.f, tf = 0.f;
  if (s == 0) { const int tb = t[b]; sa = sqrt_ac[tb]; so = sqrt_omac[tb]; }
  else if (s == 1) { tf = (float)t[b]; }
  const float coef = -logf(10000.0f) / 1023.0f;  // half-1 = 1023

  #pragma unroll
  for (int i = 0; i < 8; ++i) {
    const int d = tid + i * 256;
    float v;
    if (s == 0) {
      v = sa * x_start[(size_t)b * DIM + d] + so * noise[(size_t)b * DIM + d];
    } else if (s == 1) {
      const int h = (d < 1024) ? d : d - 1024;
      const float ang = tf * expf(coef * (float)h);
      v = (d < 1024) ? sinf(ang) : cosf(ang);
    } else if (s == 2) {
      v = cond_src[(size_t)b * DIM + d];
    } else {
      v = cond_tgt[(size_t)b * DIM + d];
    }
    vals[i] = v;
    sum += v; sumsq += v * v; sumabs += fabsf(v);
  }

  __shared__ float r0[256], r1[256], r2[256];
  r0[tid] = sum; r1[tid] = sumsq; r2[tid] = sumabs;
  __syncthreads();
  for (int sft = 128; sft > 0; sft >>= 1) {
    if (tid < sft) { r0[tid] += r0[tid+sft]; r1[tid] += r1[tid+sft]; r2[tid] += r2[tid+sft]; }
    __syncthreads();
  }
  __shared__ float s_mu, s_rstd;
  if (tid == 0) {
    const float mu  = r0[0] * (1.0f / DIM);
    const float var = fmaxf(r1[0] * (1.0f / DIM) - mu * mu, 0.f);
    s_mu = mu;
    s_rstd = rsqrtf(var + 1e-5f);
    maskf[bs] = (r2[0] > 0.f) ? 1.f : 0.f;
  }
  __syncthreads();
  const float mu = s_mu, rstd = s_rstd;
  #pragma unroll
  for (int i = 0; i < 8; ++i) {
    const int d = tid + i * 256;
    f[(size_t)bs * DIM + d] = (bf16)((vals[i] - mu) * rstd);
  }
}

// ---------------------------------------------------------------------------
// K2: fp32 -> bf16 weight conversion (3 x D x D); tile re-reads then hit L2
// ---------------------------------------------------------------------------
__global__ __launch_bounds__(256) void k_cvt_weights(
    const float* __restrict__ wq, const float* __restrict__ wk,
    const float* __restrict__ wv, bf16* __restrict__ out)
{
  const int idx   = blockIdx.x * 256 + threadIdx.x;   // < 3 * 2^22
  const int which = idx >> 22;                        // DIM*DIM = 2^22
  const int r     = idx & ((DIM * DIM) - 1);
  const float* src = (which == 0) ? wq : (which == 1) ? wk : wv;
  out[idx] = (bf16)src[r];
}

// ---------------------------------------------------------------------------
// K3: NT GEMM via WMMA bf16:  C[z] = A[M,K] * W[z][N,K]^T   (z = q,k,v)
//     block tile 128x128, BK=32, 8 waves (2x4), wave tile 64x32
//     => 8 x v_wmma_f32_16x16x32_bf16 per K-step, 512 per wave total
//     LDS fill: GLOBAL_LOAD_ASYNC_TO_LDS_B128 (ASYNCcnt), DMAing the next
//     tile into the other buffer while WMMAs run.
// ---------------------------------------------------------------------------
#define BM 128
#define BN 128
#define BK 32
#define LDA 40   // padded LDS row stride (bf16): 80B -> conflict-free, 16B aligned

__global__ __launch_bounds__(256) void k_gemm_qkv(
    const bf16* __restrict__ A,     // [M, K] row-major (f, bf16)
    const bf16* __restrict__ Wall,  // [3, N, K] row-major (bf16)
    bf16* __restrict__ Out,         // [3, M, N]
    int M, int N, int K)
{
  const bf16* __restrict__ Bw = Wall + (size_t)blockIdx.z * N * K;
  bf16* __restrict__ C        = Out  + (size_t)blockIdx.z * M * N;

  __shared__ bf16 As[2][BM * LDA];
  __shared__ bf16 Bs[2][BN * LDA];

  const int tid  = threadIdx.x;
  const int lane = tid & 31;
  const int wave = tid >> 5;
  const int wm   = wave >> 2;     // 0..1  (64 rows of M)
  const int wn   = wave & 3;      // 0..3  (32 cols of N)

  const int blockM = blockIdx.y * BM;
  const int blockN = blockIdx.x * BN;

  // cooperative fill: 512 x 16B chunks per tile, 2 per thread
  const int c0 = tid, c1 = tid + 256;
  const int r0 = c0 >> 2, cc0 = (c0 & 3) * 8;
  const int r1 = c1 >> 2, cc1 = (c1 & 3) * 8;

  v8f acc[4][2];
  const v8f vzero = {0.f,0.f,0.f,0.f,0.f,0.f,0.f,0.f};
  #pragma unroll
  for (int m = 0; m < 4; ++m)
    #pragma unroll
    for (int n = 0; n < 2; ++n) acc[m][n] = vzero;

  const int NK = K / BK;

#if USE_ASYNC_LDS
  // -------- async DMA fill path (CDNA5 GLOBAL_LOAD_ASYNC_TO_LDS_B128) ------
  auto asyncFill = [&](int buf, int kk) {
    __builtin_amdgcn_global_load_async_to_lds_b128(
        (as1_i32x4*)(as1_void*)(void*)(A  + (size_t)(blockM + r0) * K + kk + cc0),
        (as3_i32x4*)(as3_void*)&As[buf][r0 * LDA + cc0], 0, 0);
    __builtin_amdgcn_global_load_async_to_lds_b128(
        (as1_i32x4*)(as1_void*)(void*)(A  + (size_t)(blockM + r1) * K + kk + cc1),
        (as3_i32x4*)(as3_void*)&As[buf][r1 * LDA + cc1], 0, 0);
    __builtin_amdgcn_global_load_async_to_lds_b128(
        (as1_i32x4*)(as1_void*)(void*)(Bw + (size_t)(blockN + r0) * K + kk + cc0),
        (as3_i32x4*)(as3_void*)&Bs[buf][r0 * LDA + cc0], 0, 0);
    __builtin_amdgcn_global_load_async_to_lds_b128(
        (as1_i32x4*)(as1_void*)(void*)(Bw + (size_t)(blockN + r1) * K + kk + cc1),
        (as3_i32x4*)(as3_void*)&Bs[buf][r1 * LDA + cc1], 0, 0);
  };
  asyncFill(0, 0);
  wait_async0();
  __syncthreads();
#else
  // -------- fallback: VGPR staging --------
  v8bf aReg0, aReg1, bReg0, bReg1;
  auto loadTiles = [&](int kk) {
    aReg0 = *(const v8bf*)(A  + (size_t)(blockM + r0) * K + kk + cc0);
    aReg1 = *(const v8bf*)(A  + (size_t)(blockM + r1) * K + kk + cc1);
    bReg0 = *(const v8bf*)(Bw + (size_t)(blockN + r0) * K + kk + cc0);
    bReg1 = *(const v8bf*)(Bw + (size_t)(blockN + r1) * K + kk + cc1);
  };
  auto stashTiles = [&](int buf) {
    *(v8bf*)(&As[buf][r0 * LDA + cc0]) = aReg0;
    *(v8bf*)(&As[buf][r1 * LDA + cc1]) = aReg1;
    *(v8bf*)(&Bs[buf][r0 * LDA + cc0]) = bReg0;
    *(v8bf*)(&Bs[buf][r1 * LDA + cc1]) = bReg1;
  };
  loadTiles(0);
  stashTiles(0);
  __syncthreads();
#endif

  // ISA fragment geometry (16-bit A 16x32 / B 32x16, wave32)
  const int rsel = lane & 15;
  const int kloA = (lane < 16) ? 0 : 8;    // A: lanes 0-15 K{0-7,16-23}; 16-31 K{8-15,24-31}
  const int kloB = (lane < 16) ? 0 : 8;    // B: lanes 0-15 K{0-7,16-23}; 16-31 K{8-15,24-31}

  for (int kt = 0; kt < NK; ++kt) {
    const int buf = kt & 1;

#if USE_ASYNC_LDS
    if (kt + 1 < NK) asyncFill(buf ^ 1, (kt + 1) * BK);   // DMA next tile now
#else
    if (kt + 1 < NK) loadTiles((kt + 1) * BK);
#endif

    v16bf afrag[4];
    #pragma unroll
    for (int m = 0; m < 4; ++m) {
      const bf16* p = &As[buf][(wm * 64 + m * 16 + rsel) * LDA];
      v8bf lo = *(const v8bf*)(p + kloA);
      v8bf hi = *(const v8bf*)(p + kloA + 16);
      afrag[m] = __builtin_shufflevector(lo, hi, 0,1,2,3,4,5,6,7,8,9,10,11,12,13,14,15);
    }
    v16bf bfrag[2];
    #pragma unroll
    for (int n = 0; n < 2; ++n) {
      const bf16* p = &Bs[buf][(wn * 32 + n * 16 + rsel) * LDA];
      v8bf lo = *(const v8bf*)(p + kloB);
      v8bf hi = *(const v8bf*)(p + kloB + 16);
      bfrag[n] = __builtin_shufflevector(lo, hi, 0,1,2,3,4,5,6,7,8,9,10,11,12,13,14,15);
    }

    #pragma unroll
    for (int m = 0; m < 4; ++m)
      #pragma unroll
      for (int n = 0; n < 2; ++n)
        acc[m][n] = __builtin_amdgcn_wmma_f32_16x16x32_bf16(
            false, afrag[m], false, bfrag[n], (short)0, acc[m][n], false, false);

#if USE_ASYNC_LDS
    if (kt + 1 < NK) wait_async0();   // next tile resident before barrier
    __syncthreads();
#else
    __syncthreads();
    if (kt + 1 < NK) { stashTiles(buf ^ 1); __syncthreads(); }
#endif
  }

  // epilogue: C/D layout = VGPR r, lanes 0-15 -> M=r, lanes 16-31 -> M=8+r
  const int mrow_off = (lane < 16) ? 0 : 8;
  const int ncol     = lane & 15;
  #pragma unroll
  for (int m = 0; m < 4; ++m) {
    #pragma unroll
    for (int n = 0; n < 2; ++n) {
      const int gm0 = blockM + wm * 64 + m * 16 + mrow_off;
      const int gn  = blockN + wn * 32 + n * 16 + ncol;
      #pragma unroll
      for (int r = 0; r < 8; ++r)
        C[(size_t)(gm0 + r) * N + gn] = (bf16)acc[m][n][r];
    }
  }
}

// ---------------------------------------------------------------------------
// K4: per-batch 4x4 logits + masked softmax -> folded column weights
//     one wave32 per batch element (8 waves/block)
// ---------------------------------------------------------------------------
__global__ __launch_bounds__(256) void k_attn_weights(
    const bf16* __restrict__ q, const bf16* __restrict__ k,
    const float* __restrict__ maskf, float* __restrict__ colw)
{
  const int lane = threadIdx.x & 31;
  const int wave = threadIdx.x >> 5;
  const int b = blockIdx.x * 8 + wave;

  const bf16* qb = q + (size_t)b * 4 * DIM;
  const bf16* kb = k + (size_t)b * 4 * DIM;

  float acc[4][4];
  #pragma unroll
  for (int i = 0; i < 4; ++i)
    #pragma unroll
    for (int j = 0; j < 4; ++j) acc[i][j] = 0.f;

  for (int step = 0; step < 8; ++step) {
    const int e = lane * 8 + step * 256;
    v8bf qc[4], kc[4];
    #pragma unroll
    for (int i = 0; i < 4; ++i) {
      qc[i] = *(const v8bf*)(qb + i * DIM + e);
      kc[i] = *(const v8bf*)(kb + i * DIM + e);
    }
    #pragma unroll
    for (int i = 0; i < 4; ++i)
      #pragma unroll
      for (int j = 0; j < 4; ++j) {
        float p = 0.f;
        #pragma unroll
        for (int x = 0; x < 8; ++x) p += (float)qc[i][x] * (float)kc[j][x];
        acc[i][j] += p;
      }
  }

  #pragma unroll
  for (int i = 0; i < 4; ++i)
    #pragma unroll
    for (int j = 0; j < 4; ++j) {
      #pragma unroll
      for (int off = 16; off > 0; off >>= 1)
        acc[i][j] += __shfl_xor(acc[i][j], off, 32);
    }

  if (lane == 0) {
    float mk[4];
    #pragma unroll
    for (int j = 0; j < 4; ++j) mk[j] = maskf[b * 4 + j];
    const float denom = fmaxf(mk[0] + mk[1] + mk[2] + mk[3], 1.0f);
    const float scale = 0.02209708691207961f;  // 1/sqrt(2048)
    float cw[4] = {0.f, 0.f, 0.f, 0.f};
    #pragma unroll
    for (int i = 0; i < 4; ++i) {
      float lg[4]; float mx = -1e30f;
      #pragma unroll
      for (int j = 0; j < 4; ++j) {
        lg[j] = (mk[j] > 0.f) ? acc[i][j] * scale : -1e9f;
        mx = fmaxf(mx, lg[j]);
      }
      float ssum = 0.f;
      #pragma unroll
      for (int j = 0; j < 4; ++j) { lg[j] = expf(lg[j] - mx); ssum += lg[j]; }
      const float inv = (mk[i] > 0.f) ? (1.0f / ssum) : 0.f;  // query mask folds here
      #pragma unroll
      for (int j = 0; j < 4; ++j) cw[j] += lg[j] * inv;
    }
    const float idn = 1.0f / denom;
    #pragma unroll
    for (int j = 0; j < 4; ++j) colw[b * 4 + j] = cw[j] * idn;
  }
}

// ---------------------------------------------------------------------------
// K5: predicted_x0 = sum_j colw[b,j] * v[b,j,:]; per-block MSE partials
// ---------------------------------------------------------------------------
__global__ __launch_bounds__(256) void k_combine(
    const bf16* __restrict__ v, const float* __restrict__ colw,
    const float* __restrict__ x_start, float* __restrict__ out,
    float* __restrict__ partials)
{
  const int tid = threadIdx.x;
  const int idx = blockIdx.x * 256 + tid;
  const int b = idx >> 11;     // / 2048
  const int d = idx & 2047;
  const bf16* vb = v + (size_t)b * 4 * DIM + d;
  float p = 0.f;
  #pragma unroll
  for (int j = 0; j < 4; ++j) p += colw[b * 4 + j] * (float)vb[(size_t)j * DIM];
  out[1 + idx] = p;
  const float diff = p - x_start[idx];

  __shared__ float red[256];
  red[tid] = diff * diff;
  __syncthreads();
  for (int sft = 128; sft > 0; sft >>= 1) {
    if (tid < sft) red[tid] += red[tid + sft];
    __syncthreads();
  }
  if (tid == 0) partials[blockIdx.x] = red[0];
}

// ---------------------------------------------------------------------------
// K6: deterministic final reduction -> loss
// ---------------------------------------------------------------------------
__global__ __launch_bounds__(256) void k_finalize(
    const float* __restrict__ partials, int n, float* __restrict__ out)
{
  const int tid = threadIdx.x;
  float s = 0.f;
  for (int i = tid; i < n; i += 256) s += partials[i];
  __shared__ float red[256];
  red[tid] = s;
  __syncthreads();
  for (int sft = 128; sft > 0; sft >>= 1) {
    if (tid < sft) red[tid] += red[tid + sft];
    __syncthreads();
  }
  if (tid == 0) out[0] = red[0] * (1.0f / 16777216.0f);  // / (B*D)
}

// ---------------------------------------------------------------------------
extern "C" void kernel_launch(void* const* d_in, const int* in_sizes, int n_in,
                              void* d_out, int out_size, void* d_ws, size_t ws_size,
                              hipStream_t stream)
{
  (void)in_sizes; (void)n_in; (void)out_size; (void)ws_size;
  const float* x_start   = (const float*)d_in[0];
  const int*   t         = (const int*)  d_in[1];
  const float* noise     = (const float*)d_in[2];
  const float* cond_src  = (const float*)d_in[3];
  const float* cond_tgt  = (const float*)d_in[4];
  const float* w_q       = (const float*)d_in[5];
  const float* w_k       = (const float*)d_in[6];
  const float* w_v       = (const float*)d_in[7];
  const float* sqrt_ac   = (const float*)d_in[8];
  const float* sqrt_omac = (const float*)d_in[9];

  char* ws = (char*)d_ws;
  size_t off = 0;
  bf16*  f        = (bf16*) (ws + off); off += (size_t)MROWS * DIM * 2;       // 128 MB
  bf16*  wb       = (bf16*) (ws + off); off += (size_t)3 * DIM * DIM * 2;     //  24 MB
  bf16*  qkv      = (bf16*) (ws + off); off += (size_t)3 * MROWS * DIM * 2;   // 384 MB
  float* maskf    = (float*)(ws + off); off += (size_t)MROWS * 4;
  float* colw     = (float*)(ws + off); off += (size_t)MROWS * 4;
  float* partials = (float*)(ws + off); off += (size_t)65536 * 4;

  float* out = (float*)d_out;

  k_build_tokens<<<MROWS, 256, 0, stream>>>(x_start, t, noise, cond_src, cond_tgt,
                                            sqrt_ac, sqrt_omac, f, maskf);
  k_cvt_weights<<<(3 * DIM * DIM) / 256, 256, 0, stream>>>(w_q, w_k, w_v, wb);

  dim3 ggrid(DIM / BN, MROWS / BM, 3);
  k_gemm_qkv<<<ggrid, 256, 0, stream>>>(f, wb, qkv, MROWS, DIM, DIM);

  const bf16* q = qkv;
  const bf16* k = qkv + (size_t)MROWS * DIM;
  const bf16* v = qkv + (size_t)2 * MROWS * DIM;
  k_attn_weights<<<BATCH / 8, 256, 0, stream>>>(q, k, maskf, colw);
  k_combine<<<(BATCH * DIM) / 256, 256, 0, stream>>>(v, colw, x_start, out, partials);
  k_finalize<<<1, 256, 0, stream>>>(partials, (BATCH * DIM) / 256, out);
}
